// QLSTM___65481071407622
// MI455X (gfx1250) — matrix operand order repaired and checked by reading, not compiled, not run on hardware
//
#include <hip/hip_runtime.h>
#include <hip/hip_bf16.h>

// ---------------- problem constants ----------------
#define S_LEN 1024
#define BATCH 32
#define IDIM  128
#define HDIM  128
#define NHEAD 4
#define HD    32
#define ZDIM  256   // I + H
#define GDIM  512   // 4 * H (f,i,g,o concatenated)
#define NWG   16
#define TPB   256   // 8 wave32s per WG -> 128 waves total = B*NH pairs

// ---------------- workspace byte offsets (ws must be >= ~17.4 MB) ----------
#define CTR_B    0u
#define HX_B     1024u
#define CX_B     (HX_B + 16384u)
#define HXPRE_B  (CX_B + 16384u)
#define Q_B      (HXPRE_B + 16384u)
#define ATTN_B   (Q_B + 16384u)
#define GPRE_B   (ATTN_B + 16384u)                 // 32*512 f32 gate preacts
#define WGATE_B  (GPRE_B + (BATCH*GDIM*4u))        // 512x256 bf16
#define WQ_B     (WGATE_B + (GDIM*ZDIM*2u))        // 128x128 bf16
#define WK_B     (WQ_B + 32768u)
#define WV_B     (WK_B + 32768u)
#define WO_B     (WV_B + 32768u)
#define KC_B     (WO_B + 32768u)                   // [B][NH][S][HD] bf16 = 8 MB
#define VC_B     (KC_B + (32u*4u*1024u*32u*2u))    // 8 MB

typedef __attribute__((ext_vector_type(16))) __bf16 v16bf;
typedef __attribute__((ext_vector_type(8)))  float  v8f;

union FragU { uint4 u2[2]; v16bf v; };

__device__ __forceinline__ unsigned short f2bf(float f) {
  unsigned u = __builtin_bit_cast(unsigned, f);
  unsigned r = u + 0x7FFFu + ((u >> 16) & 1u);     // round-to-nearest-even
  return (unsigned short)(r >> 16);
}
__device__ __forceinline__ float bf2f(unsigned short h) {
  unsigned u = ((unsigned)h) << 16;
  return __builtin_bit_cast(float, u);
}

// A-matrix 16-bit 16x32 fragment (ISA 7.12.2 table):
// lanes 0-15: m=lane,    K = {k0..k0+7} U {k0+16..k0+23}
// lanes16-31: m=lane-16, K = {k0+8..k0+15} U {k0+24..k0+31}
__device__ __forceinline__ v16bf loadA(const unsigned short* base, int lda,
                                       int m0, int k0, int lane) {
  const unsigned short* p = base + (m0 + (lane & 15)) * lda + k0 + ((lane >> 4) << 3);
  FragU f;
  f.u2[0] = *(const uint4*)(p);
  f.u2[1] = *(const uint4*)(p + 16);
  return f.v;
}
// B-matrix 16-bit 32x16 (KxN), weights stored row-major [N][K] (W[n][k]):
// lanes 0-15: n=lane,    K = {k0..k0+7} U {k0+16..k0+23}
// lanes16-31: n=lane-16, K = {k0+8..k0+15} U {k0+24..k0+31}
__device__ __forceinline__ v16bf loadB(const unsigned short* base, int ldb,
                                       int n0, int k0, int lane) {
  const unsigned short* p = base + (n0 + (lane & 15)) * ldb + k0 + ((lane >> 4) << 3);
  FragU f;
  f.u2[0] = *(const uint4*)(p);
  f.u2[1] = *(const uint4*)(p + 16);
  return f.v;
}

// One 16x16 output tile of A(LDS, [32][K] bf16) x B(global, [N][K] bf16)^T, f32 accum
__device__ __forceinline__ v8f tile_gemm(const unsigned short* A, int lda,
                                         const unsigned short* Bm, int ldb,
                                         int m0, int n0, int K, int lane) {
  v8f c = {0.f, 0.f, 0.f, 0.f, 0.f, 0.f, 0.f, 0.f};
  for (int k0 = 0; k0 < K; k0 += 32) {
    v16bf a = loadA(A, lda, m0, k0, lane);
    v16bf b = loadB(Bm, ldb, n0, k0, lane);
    c = __builtin_amdgcn_wmma_f32_16x16x32_bf16(false, a, false, b,
                                                (short)0, c, false, false);
  }
  return c;
}

__device__ __forceinline__ float wave_max(float x) {
  #pragma unroll
  for (int off = 16; off > 0; off >>= 1) x = fmaxf(x, __shfl_xor(x, off, 32));
  return x;
}
__device__ __forceinline__ float wave_sum(float x) {
  #pragma unroll
  for (int off = 16; off > 0; off >>= 1) x += __shfl_xor(x, off, 32);
  return x;
}

// grid-wide sense-free barrier: monotonic counter, all WGs arrive, wait for epoch*NWG
__device__ __forceinline__ void grid_barrier(unsigned* ctr, unsigned epoch) {
  __syncthreads();
  if (threadIdx.x == 0) {
    __threadfence();
    __hip_atomic_fetch_add(ctr, 1u, __ATOMIC_ACQ_REL, __HIP_MEMORY_SCOPE_AGENT);
    const unsigned target = epoch * NWG;
    while (__hip_atomic_load(ctr, __ATOMIC_ACQUIRE, __HIP_MEMORY_SCOPE_AGENT) < target)
      __builtin_amdgcn_s_sleep(1);
    __threadfence();
  }
  __syncthreads();
}

// -------- init: zero state + barrier counter, convert weights f32 -> bf16 ----
__global__ void qlstm_init(const float* __restrict__ Wf, const float* __restrict__ Wi,
                           const float* __restrict__ Wg, const float* __restrict__ Wo,
                           const float* __restrict__ inw, const float* __restrict__ outw,
                           char* __restrict__ ws) {
  const size_t gid = (size_t)blockIdx.x * blockDim.x + threadIdx.x;
  const size_t gsz = (size_t)gridDim.x * blockDim.x;
  if (gid == 0) *(unsigned*)(ws + CTR_B) = 0u;
  float* hx = (float*)(ws + HX_B);
  float* cx = (float*)(ws + CX_B);
  for (size_t i = gid; i < BATCH * HDIM; i += gsz) { hx[i] = 0.f; cx[i] = 0.f; }
  unsigned short* WGATE = (unsigned short*)(ws + WGATE_B);
  for (size_t i = gid; i < (size_t)GDIM * ZDIM; i += gsz) {
    size_t r = i / ZDIM, k = i % ZDIM;
    const float* src = (r < 128) ? Wf : (r < 256) ? Wi : (r < 384) ? Wg : Wo;
    WGATE[i] = f2bf(src[(r & 127) * ZDIM + k]);
  }
  unsigned short* WQKV = (unsigned short*)(ws + WQ_B);   // WQ,WK,WV contiguous
  for (size_t i = gid; i < (size_t)3 * HDIM * HDIM; i += gsz) WQKV[i] = f2bf(inw[i]);
  unsigned short* WOp = (unsigned short*)(ws + WO_B);
  for (size_t i = gid; i < (size_t)HDIM * HDIM; i += gsz) WOp[i] = f2bf(outw[i]);
}

// -------- persistent main kernel: 1024 steps, 3 grid barriers per step -------
__global__ __launch_bounds__(TPB, 1)
void qlstm_main(const float* __restrict__ xin,
                const float* __restrict__ bf_, const float* __restrict__ bi_,
                const float* __restrict__ bg_, const float* __restrict__ bo_,
                const float* __restrict__ inb, const float* __restrict__ outb,
                float* __restrict__ out, char* __restrict__ ws) {
  __shared__ __align__(16) unsigned short zbf[BATCH * ZDIM];  // 16 KB
  __shared__ __align__(16) unsigned short hbf[BATCH * HDIM];  // 8 KB
  __shared__ __align__(16) unsigned short abf[BATCH * HDIM];  // 8 KB

  unsigned* ctr = (unsigned*)(ws + CTR_B);
  float* hx   = (float*)(ws + HX_B);
  float* cx   = (float*)(ws + CX_B);
  float* hxp  = (float*)(ws + HXPRE_B);
  float* qbuf = (float*)(ws + Q_B);
  float* ah   = (float*)(ws + ATTN_B);
  float* gpre = (float*)(ws + GPRE_B);
  const unsigned short* WGATE = (const unsigned short*)(ws + WGATE_B);
  const unsigned short* WQp = (const unsigned short*)(ws + WQ_B);
  const unsigned short* WKp = (const unsigned short*)(ws + WK_B);
  const unsigned short* WVp = (const unsigned short*)(ws + WV_B);
  const unsigned short* WOp = (const unsigned short*)(ws + WO_B);
  unsigned short* KC = (unsigned short*)(ws + KC_B);
  unsigned short* VC = (unsigned short*)(ws + VC_B);

  const int tid  = threadIdx.x;
  const int lane = tid & 31;
  const int wv   = tid >> 5;
  const int wid  = blockIdx.x * (TPB / 32) + wv;  // 0..127 = (b, head) pair
  const bool wg0 = (blockIdx.x == 0);
  const float scale = 0.17677669529663687f;       // 1/sqrt(HD)
  unsigned epoch = 0;

  for (int t = 0; t < S_LEN; ++t) {
    // ---------------- P1 (WG0): gates + LSTM pointwise + q projection -------
    if (wg0) {
      const float* xt = xin + (size_t)t * BATCH * IDIM;
      for (int i = tid; i < BATCH * ZDIM; i += TPB) {
        int b = i >> 8, k = i & 255;
        float v = (k < IDIM) ? xt[b * IDIM + k] : hx[b * HDIM + (k - IDIM)];
        zbf[i] = f2bf(v);
      }
      __syncthreads();
      // [32x256] @ [256x512]: 64 tiles over 8 waves, K=256 (8 WMMAs each)
      for (int j = 0; j < 8; ++j) {
        int tI = wv * 8 + j;
        int n0 = (tI >> 1) * 16, m0 = (tI & 1) * 16;
        v8f c = tile_gemm(zbf, ZDIM, WGATE, ZDIM, m0, n0, ZDIM, lane);
        int n  = n0 + (lane & 15);
        int mb = m0 + ((lane >> 4) << 3);
        #pragma unroll
        for (int e = 0; e < 8; ++e) gpre[(mb + e) * GDIM + n] = c[e];
      }
      __syncthreads();
      for (int i = tid; i < BATCH * HDIM; i += TPB) {
        int b = i >> 7, j = i & 127;
        const float* g = gpre + b * GDIM;
        float pf = g[j]       + bf_[j];
        float pi = g[128 + j] + bi_[j];
        float pg = g[256 + j] + bg_[j];
        float po = g[384 + j] + bo_[j];
        float f  = 1.f / (1.f + __expf(-pf));
        float ii = 1.f / (1.f + __expf(-pi));
        float gg = tanhf(pg);
        float o  = 1.f / (1.f + __expf(-po));
        float cn = f * cx[i] + ii * gg;
        float h  = o * tanhf(cn);
        cx[i] = cn; hxp[i] = h; hbf[i] = f2bf(h);
      }
      __syncthreads();
      {   // q = hx_pre @ Wq^T + bq, scale folded in
        int nt = wv;
        for (int mt = 0; mt < 2; ++mt) {
          v8f c = tile_gemm(hbf, HDIM, WQp, HDIM, mt * 16, nt * 16, HDIM, lane);
          int n  = nt * 16 + (lane & 15);
          int mb = mt * 16 + ((lane >> 4) << 3);
          #pragma unroll
          for (int e = 0; e < 8; ++e)
            qbuf[(mb + e) * HDIM + n] = (c[e] + inb[n]) * scale;
        }
      }
    }
    grid_barrier(ctr, ++epoch);   // A: hx_pre, q visible

    // ---------------- P2 (all WGs): flash attention, one (b,h) per wave -----
    if (t > 0) {
      int b = wid >> 2, h = wid & 3;
      const unsigned short* Kb = KC + (size_t)(b * NHEAD + h) * S_LEN * HD;
      const unsigned short* Vb = VC + (size_t)(b * NHEAD + h) * S_LEN * HD;
      float q = qbuf[b * HDIM + h * HD + lane];
      float m_run = -3.0e38f, l_run = 0.f, acc = 0.f;
      for (int s0 = 0; s0 < t; s0 += 32) {
        int s = s0 + lane;
        if (s0 + 32 < t)
          __builtin_prefetch((const void*)(Kb + (size_t)(s + 32) * HD), 0, 1);
        float sc = -3.0e38f;
        if (s < t) {
          const uint4* kp = (const uint4*)(Kb + (size_t)s * HD);
          uint4 ua = kp[0], ub = kp[1], uc = kp[2], ud = kp[3];
          unsigned w[16] = {ua.x, ua.y, ua.z, ua.w, ub.x, ub.y, ub.z, ub.w,
                            uc.x, uc.y, uc.z, uc.w, ud.x, ud.y, ud.z, ud.w};
          sc = 0.f;
          #pragma unroll
          for (int j2 = 0; j2 < 16; ++j2) {
            float lo = __builtin_bit_cast(float, w[j2] << 16);
            float hi = __builtin_bit_cast(float, w[j2] & 0xFFFF0000u);
            sc += __shfl(q, 2 * j2, 32) * lo + __shfl(q, 2 * j2 + 1, 32) * hi;
          }
        }
        float mc = wave_max(sc);
        float mn = fmaxf(m_run, mc);
        float sf = __expf(m_run - mn);
        float p  = __expf(sc - mn);
        l_run = l_run * sf + wave_sum(p);
        acc  *= sf;
        m_run = mn;
        int jmax = min(32, t - s0);
        const unsigned short* vrow = Vb + (size_t)s0 * HD + lane;
        for (int j2 = 0; j2 < jmax; ++j2) {
          float pj = __shfl(p, j2, 32);
          acc += pj * bf2f(vrow[(size_t)j2 * HD]);
        }
      }
      ah[b * HDIM + h * HD + lane] = acc / l_run;
    }
    grid_barrier(ctr, ++epoch);   // B: attention heads visible

    // ---------------- P3 (WG0): out-proj + residual, k/v append -------------
    if (wg0) {
      if (t > 0) {
        for (int i = tid; i < BATCH * HDIM; i += TPB) abf[i] = f2bf(ah[i]);
        __syncthreads();
        int nt = wv;
        for (int mt = 0; mt < 2; ++mt) {
          v8f c = tile_gemm(abf, HDIM, WOp, HDIM, mt * 16, nt * 16, HDIM, lane);
          int n  = nt * 16 + (lane & 15);
          int mb = mt * 16 + ((lane >> 4) << 3);
          #pragma unroll
          for (int e = 0; e < 8; ++e) {
            int m = mb + e;
            float hv = hxp[m * HDIM + n] + c[e] + outb[n];
            hx[m * HDIM + n] = hv;
            out[(size_t)t * BATCH * HDIM + m * HDIM + n] = hv;
            hbf[m * HDIM + n] = f2bf(hv);
          }
        }
      } else {
        for (int i = tid; i < BATCH * HDIM; i += TPB) {
          float hv = hxp[i];
          hx[i] = hv; out[i] = hv; hbf[i] = f2bf(hv);
        }
      }
      __syncthreads();
      // k_t / v_t projections -> append cache row t (32 tiles / 8 waves)
      for (int j = 0; j < 4; ++j) {
        int tI  = wv * 4 + j;           // 0..31
        int sel = tI >> 4;              // 0 = K, 1 = V
        int rem = tI & 15;
        int nt = rem >> 1, mt = rem & 1;
        const unsigned short* Wsel = sel ? WVp : WKp;
        const float* bsel = sel ? (inb + 2 * HDIM) : (inb + HDIM);
        unsigned short* cache = sel ? VC : KC;
        v8f c = tile_gemm(hbf, HDIM, Wsel, HDIM, mt * 16, nt * 16, HDIM, lane);
        int n  = nt * 16 + (lane & 15);
        int mb = mt * 16 + ((lane >> 4) << 3);
        int hh = n >> 5, d = n & 31;
        #pragma unroll
        for (int e = 0; e < 8; ++e) {
          int b = mb + e;
          cache[((size_t)(b * NHEAD + hh) * S_LEN + t) * HD + d] = f2bf(c[e] + bsel[n]);
        }
      }
    }
    grid_barrier(ctr, ++epoch);   // C: hx + K/V row t visible for step t+1
  }

  if (wg0) {   // tail: final hx, cx
    size_t base = (size_t)S_LEN * BATCH * HDIM;
    for (int i = tid; i < BATCH * HDIM; i += TPB) {
      out[base + i] = hx[i];
      out[base + BATCH * HDIM + i] = cx[i];
    }
  }
}

extern "C" void kernel_launch(void* const* d_in, const int* in_sizes, int n_in,
                              void* d_out, int out_size, void* d_ws, size_t ws_size,
                              hipStream_t stream) {
  (void)in_sizes; (void)n_in; (void)out_size; (void)ws_size;
  const float* xin  = (const float*)d_in[0];
  const float* Wf   = (const float*)d_in[1];
  const float* bf_  = (const float*)d_in[2];
  const float* Wi   = (const float*)d_in[3];
  const float* bi_  = (const float*)d_in[4];
  const float* Wg   = (const float*)d_in[5];
  const float* bg_  = (const float*)d_in[6];
  const float* Wo   = (const float*)d_in[7];
  const float* bo_  = (const float*)d_in[8];
  const float* inw  = (const float*)d_in[9];
  const float* inb  = (const float*)d_in[10];
  const float* outw = (const float*)d_in[11];
  const float* outb = (const float*)d_in[12];
  char* ws = (char*)d_ws;

  qlstm_init<<<64, 256, 0, stream>>>(Wf, Wi, Wg, Wo, inw, outw, ws);
  qlstm_main<<<NWG, TPB, 0, stream>>>(xin, bf_, bi_, bg_, bo_, inb, outb,
                                      (float*)d_out, ws);
}
